// GATLayer_45672682225771
// MI455X (gfx1250) — compile-verified, matrix operand order checked by reading, not run on hardware
//
#include <hip/hip_runtime.h>
#include <hip/hip_bf16.h>
#include <math.h>

#define B_   256
#define K_   64
#define D_   512
#define H_   8
#define HD_  64
#define ROWS (B_ * K_)     // 16384
#define DFF  (4 * D_)      // 2048

typedef _Float16 h2v  __attribute__((ext_vector_type(2)));
typedef _Float16 h8v  __attribute__((ext_vector_type(8)));
typedef _Float16 v16h __attribute__((ext_vector_type(16)));
typedef float    v8f  __attribute__((ext_vector_type(8)));

#define LDSP 72   // LDS row pitch in f16 (64 + 8 pad; 144B rows keep 16B alignment)

// ---------------------------------------------------------------------------
// CDNA5 async global->LDS helpers (ASYNCcnt-tracked DMA, bypasses VGPRs)
// ---------------------------------------------------------------------------
__device__ __forceinline__ unsigned lds_off(const void* p) {
    // generic pointers to LDS carry the byte offset in the low 32 bits
    return (unsigned)(unsigned long long)(uintptr_t)p;
}
__device__ __forceinline__ void async_copy16(unsigned lds_dst, const void* gsrc) {
    asm volatile("global_load_async_to_lds_b128 %0, %1, off"
                 :: "v"(lds_dst), "v"(gsrc) : "memory");
}
__device__ __forceinline__ void wait_async_le6() {   // previous 6-op group done
    asm volatile("s_wait_asynccnt 0x6" ::: "memory");
}
__device__ __forceinline__ void wait_async_0() {
    asm volatile("s_wait_asynccnt 0x0" ::: "memory");
}

// ---------------------------------------------------------------------------
// Fused LayerNorm + f32 -> f16 conversion. One block (256 thr) per row of D=512.
// ---------------------------------------------------------------------------
__global__ __launch_bounds__(256)
void ln_to_f16(const float* __restrict__ x, const float* __restrict__ g,
               const float* __restrict__ bta, _Float16* __restrict__ out) {
    __shared__ float red[256];
    const int r = blockIdx.x;
    const int t = threadIdx.x;
    const float* xr = x + (size_t)r * D_;
    float2 v = ((const float2*)xr)[t];

    red[t] = v.x + v.y;
    __syncthreads();
    for (int o = 128; o > 0; o >>= 1) { if (t < o) red[t] += red[t + o]; __syncthreads(); }
    const float mean = red[0] * (1.0f / D_);
    __syncthreads();

    const float dx = v.x - mean, dy = v.y - mean;
    red[t] = dx * dx + dy * dy;
    __syncthreads();
    for (int o = 128; o > 0; o >>= 1) { if (t < o) red[t] += red[t + o]; __syncthreads(); }
    const float rstd = rsqrtf(red[0] * (1.0f / D_) + 1e-5f);

    float2 gg = ((const float2*)g)[t];
    float2 bb = ((const float2*)bta)[t];
    h2v o2;
    o2.x = (_Float16)(dx * rstd * gg.x + bb.x);
    o2.y = (_Float16)(dy * rstd * gg.y + bb.y);
    ((h2v*)(out + (size_t)r * D_))[t] = o2;
}

// ---------------------------------------------------------------------------
// Elementwise f32 -> f16 (weights)
// ---------------------------------------------------------------------------
__global__ void cvt_f16(const float* __restrict__ in, _Float16* __restrict__ out, int n) {
    for (int i = blockIdx.x * blockDim.x + threadIdx.x; i < n; i += gridDim.x * blockDim.x)
        out[i] = (_Float16)in[i];
}

// ---------------------------------------------------------------------------
// WMMA GEMM: C[M,N] = A[M,K] * B[N,K]^T   (A,B f16 row-major, f32 accum)
// Block tile 128x64, K-step 64, double-buffered async global->LDS staging.
// 8 waves; each wave owns a 32x32 patch = 4 accumulator tiles (8 WMMA/K-step).
// MODE 0: store f32                       (h = LN(x) @ W^T)
// MODE 1: bias + exact GELU -> f16 store  (FFN up-proj)
// MODE 2: bias + residual -> f32 store    (FFN down-proj + final add)
// ---------------------------------------------------------------------------
template <int MODE>
__global__ __launch_bounds__(256)
void gemm_wmma(const _Float16* __restrict__ A, const _Float16* __restrict__ Bm,
               float* __restrict__ Cf, _Float16* __restrict__ Ch,
               const float* __restrict__ bias, const float* __restrict__ res,
               int M, int N, int Kd) {
    __shared__ __align__(16) _Float16 As[2][128 * LDSP];
    __shared__ __align__(16) _Float16 Bs[2][64 * LDSP];

    const int bm = blockIdx.y, bn = blockIdx.x;
    const int tid  = threadIdx.x;
    const int lane = tid & 31, wave = tid >> 5;
    const int wm = wave & 3;      // 32-row strip: rows wm*32 .. +31
    const int wn = wave >> 2;     // 32-col strip: cols wn*32 .. +31

    v8f acc[2][2] = {};

    const int lr = tid >> 3;              // 0..31
    const int lc = (tid & 7) * 8;         // 0,8,..,56
    const _Float16* Ab = A  + (size_t)(bm * 128) * (size_t)Kd;
    const _Float16* Bb = Bm + (size_t)(bn * 64)  * (size_t)Kd;

    // issue one 6-op async stage group: A 128x64 tile + B 64x64 tile
    auto issue = [&](int buf, int k0) {
        const unsigned abase = lds_off(&As[buf][0]);
        const unsigned bbase = lds_off(&Bs[buf][0]);
#pragma unroll
        for (int r = 0; r < 4; ++r)
            async_copy16(abase + (unsigned)(((lr + 32 * r) * LDSP + lc) * 2),
                         &Ab[(size_t)(lr + 32 * r) * Kd + k0 + lc]);
#pragma unroll
        for (int r = 0; r < 2; ++r)
            async_copy16(bbase + (unsigned)(((lr + 32 * r) * LDSP + lc) * 2),
                         &Bb[(size_t)(lr + 32 * r) * Kd + k0 + lc]);
    };

    issue(0, 0);
    int cur = 0;
    const int m = lane & 15, g = lane >> 4;

    for (int k0 = 0; k0 < Kd; k0 += 64) {
        const bool has_next = (k0 + 64) < Kd;
        if (has_next) { issue(cur ^ 1, k0 + 64); wait_async_le6(); }
        else          { wait_async_0(); }
        __syncthreads();

#pragma unroll
        for (int kk = 0; kk < 64; kk += 32) {
            union { v16h v; h8v h[2]; } a0, a1, b0, b1;
            const _Float16* ap0 = &As[cur][(wm * 32 + m) * LDSP + kk + 8 * g];
            const _Float16* ap1 = ap0 + 16 * LDSP;
            a0.h[0] = *(const h8v*)ap0;        a0.h[1] = *(const h8v*)(ap0 + 16);
            a1.h[0] = *(const h8v*)ap1;        a1.h[1] = *(const h8v*)(ap1 + 16);
            const _Float16* bp0 = &Bs[cur][(wn * 32 + m) * LDSP + kk + 8 * g];
            const _Float16* bp1 = bp0 + 16 * LDSP;
            b0.h[0] = *(const h8v*)bp0;        b0.h[1] = *(const h8v*)(bp0 + 16);
            b1.h[0] = *(const h8v*)bp1;        b1.h[1] = *(const h8v*)(bp1 + 16);
            acc[0][0] = __builtin_amdgcn_wmma_f32_16x16x32_f16(false, a0.v, false, b0.v,
                                                               (short)0, acc[0][0], false, false);
            acc[0][1] = __builtin_amdgcn_wmma_f32_16x16x32_f16(false, a0.v, false, b1.v,
                                                               (short)0, acc[0][1], false, false);
            acc[1][0] = __builtin_amdgcn_wmma_f32_16x16x32_f16(false, a1.v, false, b0.v,
                                                               (short)0, acc[1][0], false, false);
            acc[1][1] = __builtin_amdgcn_wmma_f32_16x16x32_f16(false, a1.v, false, b1.v,
                                                               (short)0, acc[1][1], false, false);
        }
        __syncthreads();   // all waves done with 'cur' before next issue overwrites it
        cur ^= 1;
    }

    // epilogue: C layout -> lane col = lane&15, VGPR v -> row v + 8*g
#pragma unroll
    for (int r = 0; r < 2; ++r)
#pragma unroll
    for (int c = 0; c < 2; ++c) {
#pragma unroll
        for (int v = 0; v < 8; ++v) {
            const int row = bm * 128 + wm * 32 + r * 16 + v + 8 * g;
            const int col = bn * 64  + wn * 32 + c * 16 + m;
            float xv = acc[r][c][v];
            if (MODE == 0) {
                Cf[(size_t)row * N + col] = xv;
            } else if (MODE == 1) {
                xv += bias[col];
                xv = 0.5f * xv * (1.0f + erff(xv * 0.70710678118654752f));
                Ch[(size_t)row * N + col] = (_Float16)xv;
            } else {
                Cf[(size_t)row * N + col] = xv + bias[col] + res[(size_t)row * N + col];
            }
        }
    }
}

// ---------------------------------------------------------------------------
// Attention per (b, head): ei/ej dots, leaky-relu + softmax, alpha @ h via WMMA,
// fused residual y = x + out. One 256-thread block per (b,h); grid = B*H.
// ---------------------------------------------------------------------------
__global__ __launch_bounds__(256)
void attn_kernel(const float* __restrict__ h, const float* __restrict__ a,
                 const float* __restrict__ x, float* __restrict__ y) {
    const int b  = blockIdx.x >> 3;      // / H_
    const int hh = blockIdx.x & 7;

    __shared__ __align__(16) _Float16 hT[64 * LDSP];   // hT[d][n] = h[b,n,hh,d]
    __shared__ __align__(16) _Float16 al[64 * LDSP];   // alpha[k][n] (f16)
    __shared__ float ei[64], ej[64];

    const int tid = threadIdx.x;
    const float* hb = h + ((size_t)b * K_ * H_ + hh) * HD_;   // + n*D_ + d

    // stage h transposed into LDS as f16 (makes both WMMA operands contiguous)
    for (int i = tid; i < 64 * 64; i += 256) {
        const int n = i >> 6, d = i & 63;
        hT[d * LDSP + n] = (_Float16)hb[(size_t)n * D_ + d];
    }
    // attention logit dot products
    if (tid < 128) {
        const int k = tid & 63;
        const float* av = a + hh * (2 * HD_) + ((tid < 64) ? 0 : HD_);
        const float* hr = hb + (size_t)k * D_;
        float s = 0.f;
        for (int d2 = 0; d2 < 64; ++d2) s += hr[d2] * av[d2];
        if (tid < 64) ei[k] = s; else ej[k] = s;
    }
    __syncthreads();

    // leaky-relu + softmax over n (f32), write alpha to LDS as f16
    if (tid < 64) {
        float ev[64];
        float mx = -1e30f;
        const float e_i = ei[tid];
        for (int n = 0; n < 64; ++n) {
            float v = e_i + ej[n];
            v = (v > 0.f) ? v : 0.2f * v;
            ev[n] = v;
            mx = fmaxf(mx, v);
        }
        float s = 0.f;
        for (int n = 0; n < 64; ++n) { float t = __expf(ev[n] - mx); ev[n] = t; s += t; }
        const float inv = 1.0f / s;
        for (int n = 0; n < 64; ++n) al[tid * LDSP + n] = (_Float16)(ev[n] * inv);
    }
    __syncthreads();

    // out[k,d] = sum_n alpha[k,n] * h[n,d]  -> A = alpha (k x n), B = hT (d x n)
    const int lane = tid & 31, wave = tid >> 5;
    const int tm  = wave & 3;
    const int tn0 = (wave >> 2) * 2;
    const int m = lane & 15, gsel = lane >> 4;
    v8f acc0 = {}; v8f acc1 = {};
#pragma unroll
    for (int kk = 0; kk < 64; kk += 32) {
        union { v16h v; h8v h[2]; } af, b0, b1;
        const _Float16* ap  = &al[(tm * 16 + m) * LDSP + kk + 8 * gsel];
        af.h[0] = *(const h8v*)ap;   af.h[1] = *(const h8v*)(ap + 16);
        const _Float16* bp0 = &hT[(tn0 * 16 + m) * LDSP + kk + 8 * gsel];
        b0.h[0] = *(const h8v*)bp0;  b0.h[1] = *(const h8v*)(bp0 + 16);
        const _Float16* bp1 = bp0 + 16 * LDSP;
        b1.h[0] = *(const h8v*)bp1;  b1.h[1] = *(const h8v*)(bp1 + 16);
        acc0 = __builtin_amdgcn_wmma_f32_16x16x32_f16(false, af.v, false, b0.v,
                                                      (short)0, acc0, false, false);
        acc1 = __builtin_amdgcn_wmma_f32_16x16x32_f16(false, af.v, false, b1.v,
                                                      (short)0, acc1, false, false);
    }

    // y = x + out (residual fused)
#pragma unroll
    for (int v = 0; v < 8; ++v) {
        const int k  = tm * 16 + v + 8 * gsel;
        const int d0 = tn0 * 16 + m;
        const size_t i0 = ((size_t)b * K_ + k) * D_ + hh * HD_ + d0;
        y[i0]      = x[i0]      + acc0[v];
        y[i0 + 16] = x[i0 + 16] + acc1[v];
    }
}

// ---------------------------------------------------------------------------
extern "C" void kernel_launch(void* const* d_in, const int* in_sizes, int n_in,
                              void* d_out, int out_size, void* d_ws, size_t ws_size,
                              hipStream_t stream) {
    const float* x      = (const float*)d_in[0];
    const float* W      = (const float*)d_in[1];
    const float* a      = (const float*)d_in[2];
    const float* norm_g = (const float*)d_in[3];
    const float* norm_b = (const float*)d_in[4];
    const float* ln_g   = (const float*)d_in[5];
    const float* ln_b   = (const float*)d_in[6];
    const float* w1     = (const float*)d_in[7];
    const float* b1     = (const float*)d_in[8];
    const float* w2     = (const float*)d_in[9];
    const float* b2     = (const float*)d_in[10];
    float* out = (float*)d_out;

    // workspace carve-up (all offsets 256B aligned)
    char* ws = (char*)d_ws;
    size_t off = 0;
    auto alloc = [&](size_t bytes) { void* p = ws + off; off = (off + bytes + 255) & ~(size_t)255; return p; };
    _Float16* xn  = (_Float16*)alloc((size_t)ROWS * D_ * 2);   // LN(x) f16; reused as LN(y)
    _Float16* Wh  = (_Float16*)alloc((size_t)D_ * D_ * 2);
    _Float16* w1h = (_Float16*)alloc((size_t)DFF * D_ * 2);
    _Float16* w2h = (_Float16*)alloc((size_t)D_ * DFF * 2);
    float*    hbuf= (float*)   alloc((size_t)ROWS * D_ * 4);
    float*    y   = (float*)   alloc((size_t)ROWS * D_ * 4);
    _Float16* zh  = (_Float16*)alloc((size_t)ROWS * DFF * 2);
    (void)ws_size; (void)n_in; (void)in_sizes; (void)out_size;

    // 0) weights -> f16
    cvt_f16<<<512, 256, 0, stream>>>(W,  Wh,  D_ * D_);
    cvt_f16<<<512, 256, 0, stream>>>(w1, w1h, DFF * D_);
    cvt_f16<<<512, 256, 0, stream>>>(w2, w2h, D_ * DFF);

    // 1) xn = LN(x) (f16)
    ln_to_f16<<<ROWS, 256, 0, stream>>>(x, norm_g, norm_b, xn);

    // 2) h = xn @ W^T  (f32)
    gemm_wmma<0><<<dim3(D_ / 64, ROWS / 128), 256, 0, stream>>>(
        xn, Wh, hbuf, nullptr, nullptr, nullptr, ROWS, D_, D_);

    // 3) attention + residual: y = x + attn(h)
    attn_kernel<<<B_ * H_, 256, 0, stream>>>(hbuf, a, x, y);

    // 4) yn = LN(y) (reuse xn buffer)
    ln_to_f16<<<ROWS, 256, 0, stream>>>(y, ln_g, ln_b, xn);

    // 5) zh = gelu(yn @ w1^T + b1) (f16)
    gemm_wmma<1><<<dim3(DFF / 64, ROWS / 128), 256, 0, stream>>>(
        xn, w1h, nullptr, zh, b1, nullptr, ROWS, DFF, D_);

    // 6) out = y + (zh @ w2^T + b2)
    gemm_wmma<2><<<dim3(D_ / 64, ROWS / 128), 256, 0, stream>>>(
        zh, w2h, out, nullptr, b2, y, ROWS, D_, DFF);
}
